// AttentionDTI_81166291960595
// MI455X (gfx1250) — matrix operand order repaired and checked by reading, not compiled, not run on hardware
//
#include <hip/hip_runtime.h>
#include <hip/hip_bf16.h>
#include <cstdint>

// ---------------------------------------------------------------------------
// AttentionDTI forward, MI455X (gfx1250), wave32 + WMMA f16->f32.
// All heavy ops (3 drug convs via implicit im2col, 2 attention einsums,
// 3 MLP layers) run through one tiled WMMA GEMM kernel with LDS-staged,
// double-buffered A tiles (async global->LDS when available).
// ---------------------------------------------------------------------------

typedef __attribute__((ext_vector_type(16))) _Float16 v16h;
typedef __attribute__((ext_vector_type(8)))  _Float16 v8h;
typedef __attribute__((ext_vector_type(8)))  float    v8f;
typedef __attribute__((ext_vector_type(4)))  int      v4i;

#define EP_RELU      0
#define EP_LEAKY     1
#define EP_RELU_FATT 2
#define EP_NONE      3

#if defined(__gfx1250__) && __has_builtin(__builtin_amdgcn_global_load_async_to_lds_b128)
#define USE_ASYNC_LDS 1
#else
#define USE_ASYNC_LDS 0
#endif

#if USE_ASYNC_LDS
typedef __attribute__((address_space(1))) v4i as1_v4i;   // global (device) AS
typedef __attribute__((address_space(3))) v4i as3_v4i;   // LDS AS
__device__ __forceinline__ void async_cp16(const void* g, void* l)
{
    // generic->AS casts laundered through integers: LDS flat addr low 32 bits
    // are the LDS byte offset; global flat == global segment address.
    __builtin_amdgcn_global_load_async_to_lds_b128(
        (as1_v4i*)(uintptr_t)g, (as3_v4i*)(uint32_t)(uintptr_t)l, 0, 0);
}
#if __has_builtin(__builtin_amdgcn_s_wait_asynccnt)
#define WAIT_ASYNC(n) __builtin_amdgcn_s_wait_asynccnt(n)
#else
#define WAIT_ASYNC(n) asm volatile("s_wait_asynccnt %0" ::"n"(n) : "memory")
#endif
#endif

// ---------------------------------------------------------------------------
// Unified WMMA GEMM:
//   Out[m, n] = act( sum_K A_row[m][K] * Bw[n, K] + bias[n] (+ fatt[b,n]) )
// A is channel-last f16 activations A[b, l, ci]; with K ordered (k, ci) the
// im2col row m = b*Lout + l is the CONTIGUOUS window starting at
// A + (b*Lin + l)*CI (length CI*KS halfs).
// Bw is f16 weights, N x K row-major (K ordered k*CI+ci).
// Workgroup: 256 thr = 8 waves; tile 128(M) x 128(N); wave = 64x32 (4x2 tiles).
// A slab (128 rows x 32 halfs per K-step) staged in LDS, double-buffered.
// Requires: M % 128 == 0, N % 128 == 0, K % 32 == 0, CI % 8 == 0.
// ---------------------------------------------------------------------------
#define LDS_ROW 40            // halfs per LDS row (32 + 8 pad -> 80B stride)
#define LDS_BUF (128 * LDS_ROW)

__global__ __launch_bounds__(256)
void gemm_wmma(const _Float16* __restrict__ A, const _Float16* __restrict__ Bw,
               const float* __restrict__ bias, _Float16* __restrict__ Out,
               const float* __restrict__ fatt,
               int Lout, int Lin, int CI, int N, int K, int mode)
{
    __shared__ _Float16 smem[2 * LDS_BUF];   // 20 KB

    const int tid  = threadIdx.x;
    const int lane = tid & 31;
    const int wave = tid >> 5;
    const int half = lane >> 4;     // 0: lanes 0-15, 1: lanes 16-31
    const int lrow = lane & 15;

    const int m_blk = blockIdx.y * 128;
    const int n_blk = blockIdx.x * 128;
    const int wm0 = (wave & 1) * 64;            // wave-local rows [wm0, wm0+64)
    const int wn0 = n_blk + (wave >> 1) * 32;   // wave cols

    // ---- cooperative A-slab loader assignment: 2 threads per row ----
    const int ldrow = tid >> 1;                 // 0..127
    const int lseg  = (tid & 1) * 2;            // segs {0,1} or {2,3} (x8 halfs)
    const _Float16* rowp;                       // contiguous K window base
    {
        const int m = m_blk + ldrow;
        const int b = m / Lout;
        const int l = m - b * Lout;
        rowp = A + ((size_t)b * Lin + l) * CI + lseg * 8;
    }
    _Float16* sdst = &smem[ldrow * LDS_ROW + lseg * 8];

#if !USE_ASYNC_LDS
    v8h pend[2][2];
#endif

    v8f acc[4][2];
#pragma unroll
    for (int i = 0; i < 4; ++i)
#pragma unroll
        for (int j = 0; j < 2; ++j)
            acc[i][j] = v8f{0.f, 0.f, 0.f, 0.f, 0.f, 0.f, 0.f, 0.f};

    const _Float16* brow[2];
#pragma unroll
    for (int nt = 0; nt < 2; ++nt)
        brow[nt] = Bw + (size_t)(wn0 + nt * 16 + lrow) * K;

    const int nsteps = K >> 5;

    // prologue: stage step 0 into buffer 0
#if USE_ASYNC_LDS
    async_cp16(rowp, sdst);
    async_cp16(rowp + 8, sdst + 8);
#else
    pend[0][0] = *(const v8h*)(rowp);
    pend[0][1] = *(const v8h*)(rowp + 8);
#endif

    for (int i = 0; i < nsteps; ++i) {
        const int kk = i << 5;
        const int cb = i & 1;
        const bool more = (i + 1) < nsteps;

        // issue next slab into the other buffer
        if (more) {
            const _Float16* rp = rowp + kk + 32;
            _Float16* sd = sdst + (1 - cb) * LDS_BUF;
#if USE_ASYNC_LDS
            async_cp16(rp, sd);
            async_cp16(rp + 8, sd + 8);
#else
            pend[1 - cb][0] = *(const v8h*)(rp);
            pend[1 - cb][1] = *(const v8h*)(rp + 8);
#endif
        }

        // make current slab visible in LDS
#if USE_ASYNC_LDS
        if (more) { WAIT_ASYNC(2); } else { WAIT_ASYNC(0); }
#else
        {
            _Float16* sd = sdst + cb * LDS_BUF;
            *(v8h*)(sd)     = pend[cb][0];
            *(v8h*)(sd + 8) = pend[cb][1];
        }
#endif
        __syncthreads();

        // B fragments: direct from global (L2-resident weights)
        v16h bfr[2];
#pragma unroll
        for (int nt = 0; nt < 2; ++nt) {
            const _Float16* p = brow[nt] + kk + half * 8; // K = half*8 + 0..7, 16 + half*8 + 0..7
            union { v16h v; v8h h[2]; } u;
            u.h[0] = *(const v8h*)p;
            u.h[1] = *(const v8h*)(p + 16);
            bfr[nt] = u.v;
        }

        // A fragments from LDS (bank-conflict-free: 80B row stride)
        const _Float16* sbuf = &smem[cb * LDS_BUF];
        v16h afr[4];
#pragma unroll
        for (int mt = 0; mt < 4; ++mt) {
            const _Float16* r = sbuf + (wm0 + mt * 16 + lrow) * LDS_ROW;
            union { v16h v; v8h h[2]; } u;
            u.h[0] = *(const v8h*)(r + half * 8);        // K = half*8 + 0..7
            u.h[1] = *(const v8h*)(r + 16 + half * 8);   // K = 16 + half*8 + 0..7
            afr[mt] = u.v;
        }

#pragma unroll
        for (int mt = 0; mt < 4; ++mt)
#pragma unroll
            for (int nt = 0; nt < 2; ++nt)
                acc[mt][nt] = __builtin_amdgcn_wmma_f32_16x16x32_f16(
                    false, afr[mt], false, bfr[nt],
                    (short)0, acc[mt][nt], false, false);

        __syncthreads();   // slab fully consumed before it is overwritten
    }

    // Epilogue: C/D layout -> lane holds N = lane%16, M = v + 8*(lane/16).
#pragma unroll
    for (int mt = 0; mt < 4; ++mt) {
#pragma unroll
        for (int nt = 0; nt < 2; ++nt) {
            const int n = wn0 + nt * 16 + lrow;
            const float bv = bias ? bias[n] : 0.f;
#pragma unroll
            for (int v = 0; v < 8; ++v) {
                const int m = m_blk + wm0 + mt * 16 + v + 8 * half;
                float val = acc[mt][nt][v] + bv;
                if (mode == EP_RELU) {
                    val = fmaxf(val, 0.f);
                } else if (mode == EP_LEAKY) {
                    val = val > 0.f ? val : 0.01f * val;
                } else if (mode == EP_RELU_FATT) {
                    const int b = m / Lout;
                    val = fmaxf(val + fatt[(size_t)b * N + n], 0.f);
                } // EP_NONE: raw
                Out[(size_t)m * N + n] = (_Float16)val;
            }
        }
    }
}

// ---------------------------------------------------------------------------
// Small prep / glue kernels.
// ---------------------------------------------------------------------------

__global__ void cvt_f16_kernel(const float* __restrict__ src,
                               _Float16* __restrict__ dst, int n)
{
    int t = blockIdx.x * blockDim.x + threadIdx.x;
    if (t < n) dst[t] = (_Float16)src[t];
}

// (CO, CI, KS) f32  ->  (CO, KS*CI) f16 with K ordered k*CI + ci.
__global__ void reorder_w_kernel(const float* __restrict__ src,
                                 _Float16* __restrict__ dst,
                                 int CO, int CI, int KS)
{
    int t = blockIdx.x * blockDim.x + threadIdx.x;
    int total = CO * CI * KS;
    if (t >= total) return;
    int k = t % KS;
    int r = t / KS;
    int ci = r % CI;
    int co = r / CI;
    dst[(size_t)co * CI * KS + (size_t)k * CI + ci] = (_Float16)src[t];
}

// x0[b, l, ci] = emb[drug[b, l], ci]  (channel-last f16)
__global__ void embed_kernel(const int* __restrict__ drug,
                             const float* __restrict__ emb,
                             _Float16* __restrict__ x0, int total)
{
    int t = blockIdx.x * blockDim.x + threadIdx.x;
    if (t >= total) return;
    int ci = t & 127;
    int bl = t >> 7;
    int tok = drug[bl];
    x0[t] = (_Float16)emb[tok * 128 + ci];
}

// Feature path: length-1 sequences with pad=1, kernel=3 -> only center tap.
// One block per batch element. Produces featureNN (f16) and
// feature_att = featureNN @ Wfa^T + bfa (f32).
__global__ __launch_bounds__(256)
void feature_kernel(const float* __restrict__ feature,
                    const float* __restrict__ fw1, const float* __restrict__ fb1,
                    const float* __restrict__ fw2, const float* __restrict__ fb2,
                    const float* __restrict__ fw3, const float* __restrict__ fb3,
                    const float* __restrict__ Wfa, const float* __restrict__ bfa,
                    _Float16* __restrict__ fNN, float* __restrict__ fatt)
{
    __shared__ float f[8];
    __shared__ float h1[128];
    __shared__ float h2[256];
    __shared__ float h3[512];
    const int b = blockIdx.x;
    const int t = threadIdx.x;

    if (t < 7) f[t] = feature[b * 7 + t];
    __syncthreads();

    if (t < 128) {
        float s = fb1[t];
        for (int i = 0; i < 7; ++i) s += fw1[(t * 7 + i) * 3 + 1] * f[i];
        h1[t] = fmaxf(s, 0.f);
    }
    __syncthreads();

    if (t < 256) {
        float s = fb2[t];
        for (int i = 0; i < 128; ++i) s += fw2[(t * 128 + i) * 3 + 1] * h1[i];
        h2[t] = fmaxf(s, 0.f);
    }
    __syncthreads();

    for (int o = t; o < 512; o += 256) {
        float s = fb3[o];
        for (int i = 0; i < 256; ++i) s += fw3[(o * 256 + i) * 3 + 1] * h2[i];
        s = fmaxf(s, 0.f);
        h3[o] = s;
        fNN[(size_t)b * 512 + o] = (_Float16)s;
    }
    __syncthreads();

    for (int d = t; d < 512; d += 256) {
        float s = bfa[d];
        const float* wrow = Wfa + (size_t)d * 512;
        for (int i = 0; i < 512; ++i) s += wrow[i] * h3[i];
        fatt[(size_t)b * 512 + d] = s;
    }
}

// Sigmoid gating + reductions:
//   vd[b,c] = max_l dconv[b,l,c] * (0.5 + sigmoid(Araw[b,l,c]))
//   vf[b,c] = fNN[b,c] * (0.5 + sigmoid(mean_l Araw[b,l,c]))
//   pair = concat(vd, vf) (f16)
__global__ void reduce_pair_kernel(const _Float16* __restrict__ dconv,
                                   const _Float16* __restrict__ araw,
                                   const _Float16* __restrict__ fNN,
                                   _Float16* __restrict__ pair,
                                   int Btot, int Ld, int Cc)
{
    int t = blockIdx.x * blockDim.x + threadIdx.x;
    if (t >= Btot * Cc) return;
    int c = t % Cc;
    int b = t / Cc;
    const size_t base = (size_t)b * Ld * Cc + c;
    float vmax = -1e30f, sum = 0.f;
    for (int l = 0; l < Ld; ++l) {
        float a = (float)araw[base + (size_t)l * Cc];
        float v = (float)dconv[base + (size_t)l * Cc];
        float comp = 1.f / (1.f + __expf(-a));
        vmax = fmaxf(vmax, v * (0.5f + comp));
        sum += a;
    }
    float feat = 1.f / (1.f + __expf(-sum / (float)Ld));
    pair[(size_t)b * 1024 + c] = (_Float16)vmax;
    pair[(size_t)b * 1024 + 512 + c] =
        (_Float16)((float)fNN[(size_t)b * Cc + c] * (0.5f + feat));
}

// Final 512 -> 1 dot product per batch element.
__global__ __launch_bounds__(128)
void final_kernel(const _Float16* __restrict__ h3, const float* __restrict__ Wo,
                  const float* __restrict__ bo, float* __restrict__ out)
{
    __shared__ float red[128];
    const int b = blockIdx.x;
    const int t = threadIdx.x;
    float s = 0.f;
    for (int i = t; i < 512; i += 128) s += (float)h3[(size_t)b * 512 + i] * Wo[i];
    red[t] = s;
    __syncthreads();
    for (int off = 64; off > 0; off >>= 1) {
        if (t < off) red[t] += red[t + off];
        __syncthreads();
    }
    if (t == 0) out[b] = red[0] + bo[0];
}

// ---------------------------------------------------------------------------
// Host launcher.
// ---------------------------------------------------------------------------
extern "C" void kernel_launch(void* const* d_in, const int* in_sizes, int n_in,
                              void* d_out, int out_size, void* d_ws, size_t ws_size,
                              hipStream_t stream)
{
    (void)in_sizes; (void)n_in; (void)out_size; (void)ws_size;

    const int Bn = 2048, LD = 100;

    const int*   drug    = (const int*)  d_in[0];
    const float* feature = (const float*)d_in[1];
    const float* emb     = (const float*)d_in[2];
    const float* dw1 = (const float*)d_in[3];  const float* db1 = (const float*)d_in[4];
    const float* dw2 = (const float*)d_in[5];  const float* db2 = (const float*)d_in[6];
    const float* dw3 = (const float*)d_in[7];  const float* db3 = (const float*)d_in[8];
    const float* fw1 = (const float*)d_in[9];  const float* fb1 = (const float*)d_in[10];
    const float* fw2 = (const float*)d_in[11]; const float* fb2 = (const float*)d_in[12];
    const float* fw3 = (const float*)d_in[13]; const float* fb3 = (const float*)d_in[14];
    const float* Wda = (const float*)d_in[15]; const float* bda = (const float*)d_in[16];
    const float* Wfa = (const float*)d_in[17]; const float* bfa = (const float*)d_in[18];
    const float* Watt= (const float*)d_in[19]; const float* batt= (const float*)d_in[20];
    const float* W1  = (const float*)d_in[21]; const float* b1  = (const float*)d_in[22];
    const float* W2  = (const float*)d_in[23]; const float* b2  = (const float*)d_in[24];
    const float* W3  = (const float*)d_in[25]; const float* b3  = (const float*)d_in[26];
    const float* Wo  = (const float*)d_in[27]; const float* bo  = (const float*)d_in[28];
    float* out = (float*)d_out;

    // ---- workspace layout (sequential, 256B aligned; total ~= 760 MB) ----
    char* cur = (char*)d_ws;
    auto alloc = [&](size_t bytes) -> char* {
        char* p = cur;
        cur += (bytes + 255) & ~(size_t)255;
        return p;
    };
    _Float16* wr1    = (_Float16*)alloc((size_t)128 * 512  * 2);
    _Float16* wr2    = (_Float16*)alloc((size_t)256 * 768  * 2);
    _Float16* wr3    = (_Float16*)alloc((size_t)512 * 2048 * 2);
    _Float16* wda16  = (_Float16*)alloc((size_t)512 * 512  * 2);
    _Float16* watt16 = (_Float16*)alloc((size_t)512 * 512  * 2);
    _Float16* w1_16  = (_Float16*)alloc((size_t)1024 * 1024 * 2);
    _Float16* w2_16  = (_Float16*)alloc((size_t)1024 * 1024 * 2);
    _Float16* w3_16  = (_Float16*)alloc((size_t)512 * 1024 * 2);
    _Float16* x0     = (_Float16*)alloc((size_t)Bn * 100 * 128 * 2);
    _Float16* act1   = (_Float16*)alloc((size_t)Bn * 97  * 128 * 2);
    _Float16* act2   = (_Float16*)alloc((size_t)Bn * 92  * 256 * 2);
    _Float16* dconv  = (_Float16*)alloc((size_t)Bn * 85  * 512 * 2);
    _Float16* s_buf  = (_Float16*)alloc((size_t)Bn * 85  * 512 * 2);
    _Float16* araw   = (_Float16*)alloc((size_t)Bn * 85  * 512 * 2);
    _Float16* fNN16  = (_Float16*)alloc((size_t)Bn * 512 * 2);
    float*    fatt   = (float*)   alloc((size_t)Bn * 512 * 4);
    _Float16* pair   = (_Float16*)alloc((size_t)Bn * 1024 * 2);
    _Float16* h1b    = (_Float16*)alloc((size_t)Bn * 1024 * 2);
    _Float16* h2b    = (_Float16*)alloc((size_t)Bn * 1024 * 2);
    _Float16* h3b    = (_Float16*)alloc((size_t)Bn * 512 * 2);

    auto blk = [](int n) { return dim3((unsigned)((n + 255) / 256)); };

    // ---- weight prep ----
    cvt_f16_kernel<<<blk(512*512),  256, 0, stream>>>(Wda,  wda16,  512*512);
    cvt_f16_kernel<<<blk(512*512),  256, 0, stream>>>(Watt, watt16, 512*512);
    cvt_f16_kernel<<<blk(1024*1024),256, 0, stream>>>(W1,   w1_16, 1024*1024);
    cvt_f16_kernel<<<blk(1024*1024),256, 0, stream>>>(W2,   w2_16, 1024*1024);
    cvt_f16_kernel<<<blk(512*1024), 256, 0, stream>>>(W3,   w3_16,  512*1024);
    reorder_w_kernel<<<blk(128*128*4), 256, 0, stream>>>(dw1, wr1, 128, 128, 4);
    reorder_w_kernel<<<blk(256*128*6), 256, 0, stream>>>(dw2, wr2, 256, 128, 6);
    reorder_w_kernel<<<blk(512*256*8), 256, 0, stream>>>(dw3, wr3, 512, 256, 8);

    // ---- embedding gather (channel-last f16) ----
    embed_kernel<<<blk(Bn*LD*128), 256, 0, stream>>>(drug, emb, x0, Bn*LD*128);

    // ---- GEMM launcher ----
    auto gemm = [&](const _Float16* A, const _Float16* Bw, const float* bias,
                    _Float16* O, const float* fa,
                    int Lout, int Lin, int CI, int KS, int N, int mode) {
        const int K = CI * KS;
        const int M = Bn * Lout;
        dim3 grid((unsigned)(N / 128), (unsigned)(M / 128));
        gemm_wmma<<<grid, 256, 0, stream>>>(A, Bw, bias, O, fa,
                                            Lout, Lin, CI, N, K, mode);
    };

    // ---- drug conv stack (implicit im2col GEMMs, ReLU fused) ----
    gemm(x0,   wr1, db1, act1,  nullptr, 97, 100, 128, 4, 128, EP_RELU);
    gemm(act1, wr2, db2, act2,  nullptr, 92, 97,  128, 6, 256, EP_RELU);
    gemm(act2, wr3, db3, dconv, nullptr, 85, 92,  256, 8, 512, EP_RELU);

    // ---- feature path (tiny; center-tap dense layers) + feature_att ----
    feature_kernel<<<dim3(Bn), 256, 0, stream>>>(feature, fw1, fb1, fw2, fb2,
                                                 fw3, fb3, Wfa, bfa, fNN16, fatt);

    // ---- attention GEMMs ----
    // s = relu(drugConv @ Wda^T + bda + feature_att)
    gemm(dconv, wda16, bda, s_buf, fatt, 85, 85, 512, 1, 512, EP_RELU_FATT);
    // Araw = s @ Watt^T + batt   (raw, sigmoids applied in reduction)
    gemm(s_buf, watt16, batt, araw, nullptr, 85, 85, 512, 1, 512, EP_NONE);

    // ---- gating + max/mean reductions -> pair ----
    reduce_pair_kernel<<<blk(Bn*512), 256, 0, stream>>>(dconv, araw, fNN16,
                                                        pair, Bn, 85, 512);

    // ---- MLP (LeakyReLU fused) ----
    gemm(pair, w1_16, b1, h1b, nullptr, 1, 1, 1024, 1, 1024, EP_LEAKY);
    gemm(h1b,  w2_16, b2, h2b, nullptr, 1, 1, 1024, 1, 1024, EP_LEAKY);
    gemm(h2b,  w3_16, b3, h3b, nullptr, 1, 1, 1024, 1,  512, EP_LEAKY);

    // ---- final 512 -> 1 ----
    final_kernel<<<dim3(Bn), 128, 0, stream>>>(h3b, Wo, bo, out);
}
